// SymbolRetrieval_69956427317858
// MI455X (gfx1250) — compile-verified, matrix operand order checked by reading, not run on hardware
//
#include <hip/hip_runtime.h>

// ---------- problem constants (from reference) ----------
#define RTOT 32768   // b*s = 4*8192 rows
#define DIN  1024    // d_in
#define BD   256     // binding_dim (keys width)
#define NS   512     // n_symbols (softmax width)
#define SD   256     // symbol_dim (output width)
#define MB   64      // rows per workgroup

typedef unsigned short u16;
typedef __attribute__((ext_vector_type(16))) __bf16    v16bf;
typedef __attribute__((ext_vector_type(8)))  float     v8f;
typedef __attribute__((ext_vector_type(4)))  float     f32x4;
typedef __attribute__((ext_vector_type(2)))  unsigned  u32x2;

// Exact RTNE f32->bf16 (used in one-time prepasses / epilogues)
static __device__ __forceinline__ u16 f2bf(float x) {
    unsigned u = __builtin_bit_cast(unsigned, x);
    u += 0x7FFFu + ((u >> 16) & 1u);
    return (u16)(u >> 16);
}

// Fast pack: 2 f32 -> packed 2xbf16 (round-half-up + v_perm_b32). lo = x, hi = y.
static __device__ __forceinline__ unsigned pack2bf(float x, float y) {
    unsigned ux = __builtin_bit_cast(unsigned, x) + 0x8000u;
    unsigned uy = __builtin_bit_cast(unsigned, y) + 0x8000u;
    // D bytes (LSB..MSB) = { ux[2], ux[3], uy[2], uy[3] }  (sel 0-3 -> s1, 4-7 -> s0)
    return __builtin_amdgcn_perm(uy, ux, 0x07060302u);
}

union Frag {
    v16bf    v;
    u16      s[16];
    unsigned w[8];
    uint4    q[2];
};

// bf16 A/B fragment from row-major [major][ld] bf16 storage (global or LDS).
// lane&15 -> row within 16-wide tile; lane>>4 -> which 8-wide K sub-chunk.
// elements 0..7 = K[k0+c .. +7], elements 8..15 = K[k0+16+c .. +7]  (c = (lane>>4)*8)
static __device__ __forceinline__ v16bf load_frag_bf16(const u16* base, int ld,
                                                       int major, int k0, int lane) {
    const u16* p = base + (size_t)(major + (lane & 15)) * ld + (k0 + ((lane >> 4) << 3));
    Frag f;
    f.q[0] = *(const uint4*)p;
    f.q[1] = *(const uint4*)(p + 16);
    return f.v;
}

// A fragment from row-major f32 storage (streamed once -> non-temporal loads),
// converted to bf16 with v_perm packing.
static __device__ __forceinline__ v16bf load_frag_f32(const float* base, int ld,
                                                      int major, int k0, int lane) {
    const float* p = base + (size_t)(major + (lane & 15)) * ld + (k0 + ((lane >> 4) << 3));
    f32x4 a0 = __builtin_nontemporal_load((const f32x4*)(p));
    f32x4 a1 = __builtin_nontemporal_load((const f32x4*)(p + 4));
    f32x4 a2 = __builtin_nontemporal_load((const f32x4*)(p + 16));
    f32x4 a3 = __builtin_nontemporal_load((const f32x4*)(p + 20));
    Frag f;
    f.w[0] = pack2bf(a0[0], a0[1]);  f.w[1] = pack2bf(a0[2], a0[3]);
    f.w[2] = pack2bf(a1[0], a1[1]);  f.w[3] = pack2bf(a1[2], a1[3]);
    f.w[4] = pack2bf(a2[0], a2[1]);  f.w[5] = pack2bf(a2[2], a2[3]);
    f.w[6] = pack2bf(a3[0], a3[1]);  f.w[7] = pack2bf(a3[2], a3[3]);
    return f.v;
}

#define WMMA_BF16(a, b, c) \
    __builtin_amdgcn_wmma_f32_16x16x32_bf16(false, (a), false, (b), (short)0, (c), false, false)

// ---------- prepass: weight conversion / transposition (L2-resident, ~1MB) ----------

// W_key [DIN][BD] f32  ->  WT [BD][DIN] bf16  (N-major, K contiguous: B-fragment friendly)
__global__ void cvt_wkeyT(const float* __restrict__ W, u16* __restrict__ WT) {
    int idx = blockIdx.x * 256 + threadIdx.x;      // BD*DIN threads
    int k = idx & (DIN - 1);
    int n = idx >> 10;                             // DIN = 1024
    WT[(size_t)n * DIN + k] = f2bf(W[(size_t)k * BD + n]);
}

// binding [NS][BD] f32 -> bf16, same layout (already K-contiguous per symbol row)
__global__ void cvt_binding(const float* __restrict__ B, u16* __restrict__ Bb) {
    int idx = blockIdx.x * 256 + threadIdx.x;      // NS*BD threads
    Bb[idx] = f2bf(B[idx]);
}

// symbols [NS][SD] f32 -> ST [SD][NS] bf16 (transpose: symbol_dim-major, j contiguous)
__global__ void cvt_symT(const float* __restrict__ S, u16* __restrict__ ST) {
    int idx = blockIdx.x * 256 + threadIdx.x;      // SD*NS threads
    int j = idx & (NS - 1);
    int n = idx >> 9;                              // NS = 512
    ST[(size_t)n * NS + j] = f2bf(S[(size_t)j * SD + n]);
}

// ---------- fused main kernel: 64 rows per block, everything in LDS ----------
// Wave tiling (8 waves): mt = wave>>1 (one 16-row M tile per wave pair),
// each wave owns 8 consecutive N tiles and keeps 8 f32 accumulators live, so the
// A fragment is loaded (and for stage A, converted) ONCE per K-step per wave.

__global__ __launch_bounds__(256) void SymbolRetrieval_fused(
        const float* __restrict__ X,   // [RTOT][DIN]
        const u16*   __restrict__ WT,  // [BD][DIN]  bf16
        const u16*   __restrict__ Bb,  // [NS][BD]   bf16
        const u16*   __restrict__ ST,  // [SD][NS]   bf16
        float*       __restrict__ Out) // [RTOT][SD]
{
    __shared__ u16   s_keys[MB * BD];     // 32 KB  bf16 keys
    __shared__ float s_scores[MB * NS];   // 128 KB f32 scores; bf16 attn overlaid in place

    const int lane = threadIdx.x & 31;
    const int wave = threadIdx.x >> 5;    // 8 waves
    const int mt   = wave >> 1;           // 0..3   (M tile of this wave)
    const int half = wave & 1;            // which half of the N tiles
    const int rowBase = blockIdx.x * MB;

    // ---- Stage A: keys = X @ W_key   (64x256, K=1024) -> LDS bf16 ----
    {
        const int nt0 = half * 8;         // 8 N-tiles per wave (16 total, width 256)
        v8f acc[8];
#pragma unroll
        for (int j = 0; j < 8; ++j) acc[j] = (v8f){0.f,0.f,0.f,0.f,0.f,0.f,0.f,0.f};
        for (int k0 = 0; k0 < DIN; k0 += 32) {
            v16bf a = load_frag_f32(X, DIN, rowBase + mt * 16, k0, lane);  // once per K-step
#pragma unroll
            for (int j = 0; j < 8; ++j) {
                v16bf b = load_frag_bf16(WT, DIN, (nt0 + j) * 16, k0, lane);
                acc[j] = WMMA_BF16(a, b, acc[j]);
            }
        }
        int mloc = mt * 16 + ((lane >> 4) << 3);
        int nsub = lane & 15;
#pragma unroll
        for (int j = 0; j < 8; ++j) {
            int n = (nt0 + j) * 16 + nsub;
#pragma unroll
            for (int v = 0; v < 8; ++v)
                s_keys[(mloc + v) * BD + n] = f2bf(acc[j][v]);
        }
    }
    __syncthreads();

    // ---- Stage B: scores = keys @ binding^T   (64x512, K=256) -> LDS f32 ----
    for (int g = 0; g < 2; ++g) {         // 16 N-tiles per wave, in 2 groups of 8
        const int nt0 = half * 16 + g * 8;
        v8f acc[8];
#pragma unroll
        for (int j = 0; j < 8; ++j) acc[j] = (v8f){0.f,0.f,0.f,0.f,0.f,0.f,0.f,0.f};
        for (int k0 = 0; k0 < BD; k0 += 32) {
            v16bf a = load_frag_bf16(s_keys, BD, mt * 16, k0, lane);
#pragma unroll
            for (int j = 0; j < 8; ++j) {
                v16bf b = load_frag_bf16(Bb, BD, (nt0 + j) * 16, k0, lane);
                acc[j] = WMMA_BF16(a, b, acc[j]);
            }
        }
        int mloc = mt * 16 + ((lane >> 4) << 3);
        int nsub = lane & 15;
#pragma unroll
        for (int j = 0; j < 8; ++j) {
            int n = (nt0 + j) * 16 + nsub;
#pragma unroll
            for (int v = 0; v < 8; ++v)
                s_scores[(mloc + v) * NS + n] = acc[j][v];
        }
    }
    __syncthreads();

    // ---- Stage C: row softmax over 512 (vectorized b128 LDS reads),
    //      bf16 attn written in place via packed b64 stores (wave-private rows) ----
    for (int r = wave * 8; r < wave * 8 + 8; ++r) {
        float* srow = s_scores + r * NS;
        f32x4 v[4];
#pragma unroll
        for (int j = 0; j < 4; ++j)
            v[j] = *(const f32x4*)(srow + lane * 4 + j * 128);
        float m = v[0][0];
#pragma unroll
        for (int j = 0; j < 4; ++j)
#pragma unroll
            for (int e = 0; e < 4; ++e) m = fmaxf(m, v[j][e]);
#pragma unroll
        for (int off = 16; off >= 1; off >>= 1) m = fmaxf(m, __shfl_xor(m, off, 32));
        float ssum = 0.f;
#pragma unroll
        for (int j = 0; j < 4; ++j)
#pragma unroll
            for (int e = 0; e < 4; ++e) { v[j][e] = __expf(v[j][e] - m); ssum += v[j][e]; }
#pragma unroll
        for (int off = 16; off >= 1; off >>= 1) ssum += __shfl_xor(ssum, off, 32);
        float inv = 1.0f / ssum;
        unsigned* arow = (unsigned*)srow;     // overlay: row's own storage, no cross-wave race
#pragma unroll
        for (int j = 0; j < 4; ++j) {
            u32x2 pkt;
            pkt[0] = pack2bf(v[j][0] * inv, v[j][1] * inv);
            pkt[1] = pack2bf(v[j][2] * inv, v[j][3] * inv);
            *(u32x2*)(arow + j * 64 + lane * 2) = pkt;   // ds_store_b64
        }
    }
    __syncthreads();

    // ---- Stage D: out = attn @ symbols   (64x256, K=512) -> global f32 (NT stores) ----
    {
        const u16* s_attn = (const u16*)s_scores;  // row stride = 2*NS u16 (f32 row footprint)
        const int nt0 = half * 8;
        v8f acc[8];
#pragma unroll
        for (int j = 0; j < 8; ++j) acc[j] = (v8f){0.f,0.f,0.f,0.f,0.f,0.f,0.f,0.f};
        for (int k0 = 0; k0 < NS; k0 += 32) {
            v16bf a = load_frag_bf16(s_attn, 2 * NS, mt * 16, k0, lane);
#pragma unroll
            for (int j = 0; j < 8; ++j) {
                v16bf b = load_frag_bf16(ST, NS, (nt0 + j) * 16, k0, lane);
                acc[j] = WMMA_BF16(a, b, acc[j]);
            }
        }
        int mloc = mt * 16 + ((lane >> 4) << 3);
        int nsub = lane & 15;
#pragma unroll
        for (int j = 0; j < 8; ++j) {
            int n = (nt0 + j) * 16 + nsub;
#pragma unroll
            for (int v = 0; v < 8; ++v)
                __builtin_nontemporal_store(acc[j][v],
                    Out + (size_t)(rowBase + mloc + v) * SD + n);
        }
    }
}

// ---------- host-side launcher ----------
extern "C" void kernel_launch(void* const* d_in, const int* in_sizes, int n_in,
                              void* d_out, int out_size, void* d_ws, size_t ws_size,
                              hipStream_t stream) {
    const float* X   = (const float*)d_in[0];   // [4,8192,1024]
    const float* Wk  = (const float*)d_in[1];   // [1024,256]
    const float* Bnd = (const float*)d_in[2];   // [512,256]
    const float* Sym = (const float*)d_in[3];   // [512,256]
    float* Out = (float*)d_out;                 // [4,8192,256]

    // workspace layout (bf16 weights): WT 512KB | binding 256KB | symT 256KB
    u16* WT = (u16*)d_ws;
    u16* Bb = WT + (size_t)BD * DIN;
    u16* ST = Bb + (size_t)NS * BD;

    cvt_wkeyT  <<<(BD * DIN) / 256, 256, 0, stream>>>(Wk, WT);
    cvt_binding<<<(NS * BD) / 256, 256, 0, stream>>>(Bnd, Bb);
    cvt_symT   <<<(SD * NS) / 256, 256, 0, stream>>>(Sym, ST);

    SymbolRetrieval_fused<<<RTOT / MB, 256, 0, stream>>>(X, WT, Bb, ST, Out);
}
